// MoE_56418690400539
// MI455X (gfx1250) — compile-verified, matrix operand order checked by reading, not run on hardware
//
#include <hip/hip_runtime.h>

// MoE: B=8, S=4096, D=512, H=2048, E=8, K=2  (all fp32 in/out)
#define NTOK 32768
#define Dk   512
#define Hk   2048
#define Ek   8
#define MTOK 32              // tokens per expert block (2 WMMA M-tiles)
#define HC   512             // H chunk held in LDS per pass (4 chunks)
#define XLD  520             // padded LDS row stride (halves): 1040B = 65*16
#define HLD  520             // padded LDS row stride (halves) for h chunk

typedef _Float16 v16h __attribute__((ext_vector_type(16)));
typedef _Float16 v8h  __attribute__((ext_vector_type(8)));
typedef float    v8f  __attribute__((ext_vector_type(8)));

static __device__ inline v16h cat16(v8h lo, v8h hi) {
  return __builtin_shufflevector(lo, hi, 0,1,2,3,4,5,6,7,8,9,10,11,12,13,14,15);
}

// ---------------------------------------------------------------------------
// Pre-swizzle [Kd x Nd] f32 row-major matrices into the per-lane WMMA-B f16
// fragment layout: tile = (mat*(Kd/32)+kb)*(Nd/16)+nb, 512 halves per tile,
// lane-contiguous. B fragment load becomes one 32B/lane load, no cvt VALU.
// ---------------------------------------------------------------------------
template <int Kd, int Nd>
__global__ __launch_bounds__(256) void swizzleB_kernel(
    const float* __restrict__ src, _Float16* __restrict__ dst)
{
  const size_t gid = (size_t)blockIdx.x * 256 + threadIdx.x;  // tile*32 + lane
  const int lane = (int)(gid & 31);
  const size_t tile = gid >> 5;
  constexpr int NTN = Nd >> 4;   // n-tiles
  constexpr int NTK = Kd >> 5;   // k-tiles
  const int nb = (int)(tile % NTN);
  const size_t rest = tile / NTN;
  const int kb = (int)(rest % NTK);
  const size_t mat = rest / NTK;

  const int n   = nb * 16 + (lane & 15);
  const int hi8 = (lane >> 4) << 3;
  const float* s = src + (size_t)mat * Kd * Nd;
  v16h o;
#pragma unroll
  for (int i = 0; i < 8; ++i)
    o[i] = (_Float16)s[(size_t)(kb * 32 + hi8 + i) * Nd + n];
#pragma unroll
  for (int i = 0; i < 8; ++i)
    o[8 + i] = (_Float16)s[(size_t)(kb * 32 + 16 + hi8 + i) * Nd + n];
  *(v16h*)(dst + gid * 16) = o;
}

// ---------------------------------------------------------------------------
// Router: one wave32 per token. Dual dot products (Wg, Wn), softplus-noised
// logits, top-2, analytic 2-way softmax gates, append to per-expert lists.
// ---------------------------------------------------------------------------
__global__ __launch_bounds__(256) void router_kernel(
    const float* __restrict__ x, const float* __restrict__ noise,
    const float* __restrict__ Wg, const float* __restrict__ bg,
    const float* __restrict__ Wn, const float* __restrict__ bn,
    int* __restrict__ cnt, int* __restrict__ tok_list, float* __restrict__ gate_list)
{
  const int wave = threadIdx.x >> 5;
  const int lane = threadIdx.x & 31;
  const int t = blockIdx.x * 8 + wave;

  float ag[Ek], an[Ek];
#pragma unroll
  for (int e = 0; e < Ek; ++e) { ag[e] = 0.f; an[e] = 0.f; }

  const float* xt = x + (size_t)t * Dk;
  for (int d = lane; d < Dk; d += 32) {
    const float xv = xt[d];
    const float* wg = Wg + d * Ek;
    const float* wn = Wn + d * Ek;
#pragma unroll
    for (int e = 0; e < Ek; ++e) {
      ag[e] = fmaf(xv, wg[e], ag[e]);
      an[e] = fmaf(xv, wn[e], an[e]);
    }
  }
#pragma unroll
  for (int off = 16; off > 0; off >>= 1) {
#pragma unroll
    for (int e = 0; e < Ek; ++e) {
      ag[e] += __shfl_down(ag[e], off, 32);
      an[e] += __shfl_down(an[e], off, 32);
    }
  }

  if (lane == 0) {
    float v[Ek];
#pragma unroll
    for (int e = 0; e < Ek; ++e) {
      const float lg = ag[e] + bg[e];
      const float ln = an[e] + bn[e];
      const float sp = (ln > 0.f) ? (ln + log1pf(expf(-ln))) : log1pf(expf(ln));
      v[e] = fmaf(noise[(size_t)t * Ek + e], sp, lg);
    }
    int i1 = 0;
#pragma unroll
    for (int e = 1; e < Ek; ++e) if (v[e] > v[i1]) i1 = e;
    int i2 = (i1 == 0) ? 1 : 0;
#pragma unroll
    for (int e = 0; e < Ek; ++e) if (e != i1 && v[e] > v[i2]) i2 = e;
    const float r  = expf(v[i2] - v[i1]);
    const float g1 = 1.f / (1.f + r);
    const float g2 = r / (1.f + r);
    int p1 = atomicAdd(&cnt[i1], 1);
    tok_list[i1 * NTOK + p1]  = t;
    gate_list[i1 * NTOK + p1] = g1;
    int p2 = atomicAdd(&cnt[i2], 1);
    tok_list[i2 * NTOK + p2]  = t;
    gate_list[i2 * NTOK + p2] = g2;
  }
}

// ---------------------------------------------------------------------------
// Expert MLP: block = (32 routed tokens) x (1 expert); 8 waves.
// Per k-step: two A fragments (M-tiles) from LDS; each streamed B fragment
// (one 32B/lane load from pre-swizzled f16 weights) feeds two WMMAs.
// H processed in 4 chunks of 512 so h stays in LDS; y accumulated in VGPRs.
// ---------------------------------------------------------------------------
__global__ __launch_bounds__(256, 1) void expert_kernel(
    const float* __restrict__ x,
    const _Float16* __restrict__ W1s, const float* __restrict__ b1,
    const _Float16* __restrict__ W2s, const float* __restrict__ b2,
    const int* __restrict__ cnt, const int* __restrict__ tok_list,
    const float* __restrict__ gate_list, float* __restrict__ out)
{
  __shared__ _Float16 xs[MTOK * XLD];   // 32 tokens x 512 (f16, padded)
  __shared__ _Float16 hs[MTOK * HLD];   // 32 tokens x 512 (f16, padded)
  __shared__ int   stok[MTOK];
  __shared__ float sgate[MTOK];

  const int e  = blockIdx.y;
  const int m0 = blockIdx.x * MTOK;
  const int ce = cnt[e];
  if (m0 >= ce) return;
  const int rows = (ce - m0 < MTOK) ? (ce - m0) : MTOK;

  const int tid = threadIdx.x;
  if (tid < MTOK) {
    const bool valid = tid < rows;
    stok[tid]  = valid ? tok_list[e * NTOK + m0 + tid] : 0;
    sgate[tid] = valid ? gate_list[e * NTOK + m0 + tid] : 0.f;
  }
  __syncthreads();

  // Gather 32 token rows of x into LDS as f16 (coalesced per row).
  for (int i = tid; i < MTOK * Dk; i += 256) {
    const int r = i >> 9, c = i & (Dk - 1);
    const float v = (r < rows) ? x[(size_t)stok[r] * Dk + c] : 0.f;
    xs[r * XLD + c] = (_Float16)v;
  }
  __syncthreads();

  const int wave = tid >> 5;
  const int lane = tid & 31;
  const int ln   = lane & 15;          // A row / B,C,D column within tile
  const int hi8  = (lane >> 4) << 3;   // 0 or 8: K/M offset for upper lanes

  const float* b1e = b1 + e * Hk;
  const float* b2e = b2 + e * Dk;

  v8f yacc[4][2];
#pragma unroll
  for (int q = 0; q < 4; ++q) { yacc[q][0] = (v8f){}; yacc[q][1] = (v8f){}; }

  for (int ch = 0; ch < Hk; ch += HC) {
    // ---- up-projection: hs = relu(xs @ W1e[:, ch:ch+HC] + b1) ----
    const int nb0 = (ch >> 4) + wave * 4;      // first n-tile (of 128) for wave
    v8f acc[4][2];
#pragma unroll
    for (int p = 0; p < 4; ++p) { acc[p][0] = (v8f){}; acc[p][1] = (v8f){}; }

    for (int k0 = 0; k0 < Dk; k0 += 32) {
      v16h A[2];
#pragma unroll
      for (int mt = 0; mt < 2; ++mt) {
        const int r = (mt * 16 + ln) * XLD;
        A[mt] = cat16(*(const v8h*)&xs[r + k0 + hi8],
                      *(const v8h*)&xs[r + k0 + 16 + hi8]);
      }
      // swizzled tile base: tile = (e*16 + k0/32)*128 + nb
      const _Float16* bK = W1s +
          ((size_t)((e * 16 + (k0 >> 5)) * 128 + nb0)) * 512 + lane * 16;
      __builtin_prefetch((const void*)(bK + (size_t)128 * 512), 0, 3);
#pragma unroll
      for (int p = 0; p < 4; ++p) {
        const v16h B = *(const v16h*)(bK + (size_t)p * 512);
        acc[p][0] = __builtin_amdgcn_wmma_f32_16x16x32_f16(
            false, A[0], false, B, (short)0, acc[p][0], false, false);
        acc[p][1] = __builtin_amdgcn_wmma_f32_16x16x32_f16(
            false, A[1], false, B, (short)0, acc[p][1], false, false);
      }
    }
    // epilogue: bias + relu + f16 store into hs
#pragma unroll
    for (int p = 0; p < 4; ++p) {
      const int ncol = (nb0 + p) * 16 + ln;    // global column in H
      const float bias = b1e[ncol];
      const int nloc = ncol - ch;              // 0..HC-1
#pragma unroll
      for (int mt = 0; mt < 2; ++mt) {
#pragma unroll
        for (int vv = 0; vv < 8; ++vv) {
          const int m = mt * 16 + vv + hi8;    // C/D row layout
          hs[m * HLD + nloc] = (_Float16)fmaxf(acc[p][mt][vv] + bias, 0.f);
        }
      }
    }
    __syncthreads();

    // ---- down-projection partial: yacc += hs @ W2e[ch:ch+HC, :] ----
    const int db0 = wave * 4;                  // first n-tile (of 32) for wave
    for (int k0 = 0; k0 < HC; k0 += 32) {
      v16h A[2];
#pragma unroll
      for (int mt = 0; mt < 2; ++mt) {
        const int r = (mt * 16 + ln) * HLD;
        A[mt] = cat16(*(const v8h*)&hs[r + k0 + hi8],
                      *(const v8h*)&hs[r + k0 + 16 + hi8]);
      }
      const int kbg = (ch + k0) >> 5;          // global k-block in H
      const _Float16* bK = W2s +
          ((size_t)((e * 64 + kbg) * 32 + db0)) * 512 + lane * 16;
      __builtin_prefetch((const void*)(bK + (size_t)32 * 512), 0, 3);
#pragma unroll
      for (int q = 0; q < 4; ++q) {
        const v16h B = *(const v16h*)(bK + (size_t)q * 512);
        yacc[q][0] = __builtin_amdgcn_wmma_f32_16x16x32_f16(
            false, A[0], false, B, (short)0, yacc[q][0], false, false);
        yacc[q][1] = __builtin_amdgcn_wmma_f32_16x16x32_f16(
            false, A[1], false, B, (short)0, yacc[q][1], false, false);
      }
    }
    __syncthreads();
  }

  // ---- scatter: out[token] += (y + b2) * gate ----
#pragma unroll
  for (int q = 0; q < 4; ++q) {
    const int ncol = (wave * 4 + q) * 16 + ln;
    const float bias = b2e[ncol];
#pragma unroll
    for (int mt = 0; mt < 2; ++mt) {
#pragma unroll
      for (int vv = 0; vv < 8; ++vv) {
        const int m = mt * 16 + vv + hi8;
        if (m < rows) {
          const float val = (yacc[q][mt][vv] + bias) * sgate[m];
          atomicAdd(&out[(size_t)stok[m] * Dk + ncol], val);
        }
      }
    }
  }
}

// ---------------------------------------------------------------------------
extern "C" void kernel_launch(void* const* d_in, const int* in_sizes, int n_in,
                              void* d_out, int out_size, void* d_ws, size_t ws_size,
                              hipStream_t stream) {
  (void)in_sizes; (void)n_in; (void)ws_size;
  const float* x     = (const float*)d_in[0];
  const float* noise = (const float*)d_in[1];
  const float* Wg    = (const float*)d_in[2];
  const float* bg    = (const float*)d_in[3];
  const float* Wn    = (const float*)d_in[4];
  const float* bn    = (const float*)d_in[5];
  const float* W1    = (const float*)d_in[6];
  const float* b1    = (const float*)d_in[7];
  const float* W2    = (const float*)d_in[8];
  const float* b2    = (const float*)d_in[9];
  float* out = (float*)d_out;

  // Workspace layout:
  //   [cnt: 256B] [tok_list: E*NTOK*4 = 1MB] [gate_list: 1MB]
  //   [W1s f16 swizzled: E*D*H*2 = 16MB] [W2s f16 swizzled: 16MB]
  char* ws = (char*)d_ws;
  int*   cnt  = (int*)ws;
  int*   tokl = (int*)(ws + 256);
  float* gatl = (float*)(ws + 256 + (size_t)Ek * NTOK * sizeof(int));
  _Float16* W1s = (_Float16*)(ws + 4 * 1024 * 1024);
  _Float16* W2s = W1s + (size_t)Ek * Dk * Hk;

  hipMemsetAsync(out, 0, (size_t)out_size * sizeof(float), stream);
  hipMemsetAsync(cnt, 0, Ek * sizeof(int), stream);

  // Weight pre-swizzle into WMMA-B fragment layout (f32 -> f16).
  {
    const size_t g1 = (size_t)Ek * (Dk / 32) * (Hk / 16) * 32;   // W1 lanes
    const size_t g2 = (size_t)Ek * (Hk / 32) * (Dk / 16) * 32;   // W2 lanes
    swizzleB_kernel<Dk, Hk><<<(int)(g1 / 256), 256, 0, stream>>>(W1, W1s);
    swizzleB_kernel<Hk, Dk><<<(int)(g2 / 256), 256, 0, stream>>>(W2, W2s);
  }

  router_kernel<<<NTOK / 8, 256, 0, stream>>>(x, noise, Wg, bg, Wn, bn,
                                              cnt, tokl, gatl);
  expert_kernel<<<dim3(NTOK / MTOK, Ek), 256, 0, stream>>>(x, W1s, b1, W2s, b2,
                                                           cnt, tokl, gatl, out);
}
